// CoattentionNet_89232240541934
// MI455X (gfx1250) — matrix operand order repaired and checked by reading, not compiled, not run on hardware
//
#include <hip/hip_runtime.h>

// ---------------------------------------------------------------------------
// CDNA5 (gfx1250) co-attention pipeline, bf16 WMMA (V_WMMA_F32_16X16X32_BF16)
// Heavy GEMMs software-pipelined (double-buffered fragments).
// ---------------------------------------------------------------------------

typedef __bf16 bf16_t;
typedef bf16_t v16bf __attribute__((ext_vector_type(16)));
typedef bf16_t v8bf  __attribute__((ext_vector_type(8)));
typedef float  v8f   __attribute__((ext_vector_type(8)));

constexpr int NB = 64;    // batch
constexpr int ND = 512;   // embed dim d
constexpr int NM = 1024;  // M
constexpr int NL = 256;   // L
constexpr int NK = 30;    // attention heads k
constexpr int NKP = 32;   // padded k

// ---- bf16 helpers ---------------------------------------------------------
__device__ __forceinline__ unsigned short f32_to_bf16_bits(float f) {
  unsigned u = __builtin_bit_cast(unsigned, f);
  unsigned r = u + 0x7fffu + ((u >> 16) & 1u);   // round-to-nearest-even
  return (unsigned short)(r >> 16);
}
__device__ __forceinline__ bf16_t f32_to_bf16(float f) {
  return __builtin_bit_cast(bf16_t, f32_to_bf16_bits(f));
}
__device__ __forceinline__ unsigned pack2(float a, float b) {
  return (unsigned)f32_to_bf16_bits(a) | ((unsigned)f32_to_bf16_bits(b) << 16);
}

// A fragment (16x32 bf16): lane hi-half selects kb=16*hi; 16 contiguous k
__device__ __forceinline__ v16bf fragA(const bf16_t* p) {
  union { v16bf v; v8bf h[2]; } u;
  u.h[0] = *(const v8bf*)(p);
  u.h[1] = *(const v8bf*)(p + 8);
  return u.v;
}
// B fragment (32x16 bf16): lane hi-half selects kb=16*hi; 16 contiguous k
__device__ __forceinline__ v16bf fragB(const bf16_t* p) {
  union { v16bf v; v8bf h[2]; } u;
  u.h[0] = *(const v8bf*)(p);
  u.h[1] = *(const v8bf*)(p + 8);
  return u.v;
}
__device__ __forceinline__ v8f wmma_bf16(v16bf a, v16bf b, v8f c) {
  return __builtin_amdgcn_wmma_f32_16x16x32_bf16(false, a, false, b, (short)0, c,
                                                 false, false);
}

// ---- conversion / layout kernels ------------------------------------------
__global__ void coatt_convert_bf16(const float* __restrict__ src,
                                   bf16_t* __restrict__ dst, int n) {
  for (int i = blockIdx.x * blockDim.x + threadIdx.x; i < n;
       i += gridDim.x * blockDim.x)
    dst[i] = f32_to_bf16(src[i]);
}

__global__ void coatt_pad_weights(const float* __restrict__ Wv,
                                  const float* __restrict__ Wq,
                                  const float* __restrict__ whv,
                                  const float* __restrict__ whq,
                                  bf16_t* __restrict__ Wvp, bf16_t* __restrict__ Wqp,
                                  float* __restrict__ whvp, float* __restrict__ whqp) {
  int i = blockIdx.x * blockDim.x + threadIdx.x;   // 0 .. 32*512-1
  if (i >= NKP * ND) return;
  int r = i / ND, c = i % ND;
  float v = (r < NK) ? Wv[r * ND + c] : 0.f;
  float q = (r < NK) ? Wq[r * ND + c] : 0.f;
  Wvp[i] = f32_to_bf16(v);
  Wqp[i] = f32_to_bf16(q);
  if (i < NKP) {
    whvp[i] = (i < NK) ? whv[i] : 0.f;
    whqp[i] = (i < NK) ? whq[i] : 0.f;
  }
}

// V [B, d, M] f32 -> Vt [B, M, d] bf16 (LDS-tiled transpose)
__global__ void coatt_transpose_V(const float* __restrict__ V,
                                  bf16_t* __restrict__ Vt) {
  __shared__ float tile[32][33];
  int b = blockIdx.z;
  int m0 = blockIdx.x * 32, d0 = blockIdx.y * 32;
  int tx = threadIdx.x, ty = threadIdx.y;
  const float* Vb = V + (size_t)b * ND * NM;
#pragma unroll
  for (int i = 0; i < 4; i++)
    tile[ty + 8 * i][tx] = Vb[(size_t)(d0 + ty + 8 * i) * NM + m0 + tx];
  __syncthreads();
  bf16_t* Vtb = Vt + (size_t)b * NM * ND;
#pragma unroll
  for (int i = 0; i < 4; i++)
    Vtb[(size_t)(m0 + ty + 8 * i) * ND + d0 + tx] = f32_to_bf16(tile[tx][ty + 8 * i]);
}

// ---- pipelined 16x64-tile bf16 GEMM core (K = ND), 4 accumulators ---------
// arow: A row base (+kbA); brow[j]: Bt row bases (+kbB). Returns acc[0..3].
__device__ __forceinline__ void gemm_core_pipe(const bf16_t* __restrict__ arow,
                                               const bf16_t* const brow[4],
                                               v8f acc[4]) {
  v16bf a_cur = fragA(arow);
  v16bf b_cur0 = fragB(brow[0]);
  v16bf b_cur1 = fragB(brow[1]);
  v16bf b_cur2 = fragB(brow[2]);
  v16bf b_cur3 = fragB(brow[3]);
  for (int k0 = 0; k0 < ND - 32; k0 += 32) {
    // prefetch next k-step into fresh registers (no full loadcnt drain)
    v16bf a_nxt = fragA(arow + k0 + 32);
    v16bf b_nxt0 = fragB(brow[0] + k0 + 32);
    v16bf b_nxt1 = fragB(brow[1] + k0 + 32);
    v16bf b_nxt2 = fragB(brow[2] + k0 + 32);
    v16bf b_nxt3 = fragB(brow[3] + k0 + 32);
    acc[0] = wmma_bf16(a_cur, b_cur0, acc[0]);
    acc[1] = wmma_bf16(a_cur, b_cur1, acc[1]);
    acc[2] = wmma_bf16(a_cur, b_cur2, acc[2]);
    acc[3] = wmma_bf16(a_cur, b_cur3, acc[3]);
    a_cur = a_nxt;
    b_cur0 = b_nxt0; b_cur1 = b_nxt1; b_cur2 = b_nxt2; b_cur3 = b_nxt3;
  }
  acc[0] = wmma_bf16(a_cur, b_cur0, acc[0]);
  acc[1] = wmma_bf16(a_cur, b_cur1, acc[1]);
  acc[2] = wmma_bf16(a_cur, b_cur2, acc[2]);
  acc[3] = wmma_bf16(a_cur, b_cur3, acc[3]);
}

// ---- GEMM 1: WbV = W_b @ V ; stored transposed as WbVt[b][m][d] bf16 -------
__global__ void coatt_gemm_wbv(const bf16_t* __restrict__ Wbb,
                               const bf16_t* __restrict__ Vt,
                               bf16_t* __restrict__ WbVt) {
  const int wpB = (ND / 16) * (NM / 64);          // 512 waves/batch
  int w = (blockIdx.x * blockDim.x + threadIdx.x) >> 5;
  int lane = threadIdx.x & 31;
  int b = w / wpB;
  if (b >= NB) return;
  int t = w % wpB;
  int tm = t / (NM / 64);                          // d-row tile  (0..31)
  int tn = t % (NM / 64);                          // m 64-tile   (0..15)
  int q = lane & 15, hi = lane >> 4;
  int kbA = hi * 16, kbB = hi * 16;

  const bf16_t* arow = Wbb + (size_t)(tm * 16 + q) * ND + kbA;
  const bf16_t* Vtb = Vt + (size_t)b * NM * ND;
  const bf16_t* brow[4];
#pragma unroll
  for (int j = 0; j < 4; j++)
    brow[j] = Vtb + (size_t)(tn * 64 + j * 16 + q) * ND + kbB;

  v8f acc[4] = {};
  gemm_core_pipe(arow, brow, acc);

  bf16_t* Ob = WbVt + (size_t)b * NM * ND;
  int d0 = tm * 16 + hi * 8;
#pragma unroll
  for (int j = 0; j < 4; j++) {
    int m = tn * 64 + j * 16 + q;
    uint4 pk;
    pk.x = pack2(acc[j][0], acc[j][1]);
    pk.y = pack2(acc[j][2], acc[j][3]);
    pk.z = pack2(acc[j][4], acc[j][5]);
    pk.w = pack2(acc[j][6], acc[j][7]);
    *(uint4*)(Ob + (size_t)m * ND + d0) = pk;
  }
}

// ---- GEMM 2: C = Q @ WbV ; stored as Cml[b][m][l] and Clm[b][l][m] bf16 ----
__global__ void coatt_gemm_affinity(const bf16_t* __restrict__ Qb,
                                    const bf16_t* __restrict__ WbVt,
                                    bf16_t* __restrict__ Cml,
                                    unsigned short* __restrict__ Clm) {
  const int wpB = (NL / 16) * (NM / 64);          // 256 waves/batch
  int w = (blockIdx.x * blockDim.x + threadIdx.x) >> 5;
  int lane = threadIdx.x & 31;
  int b = w / wpB;
  if (b >= NB) return;
  int t = w % wpB;
  int tm = t / (NM / 64);                          // l tile (0..15)
  int tn = t % (NM / 64);                          // m 64-tile (0..15)
  int q = lane & 15, hi = lane >> 4;
  int kbA = hi * 16, kbB = hi * 16;

  const bf16_t* arow = Qb + ((size_t)b * NL + tm * 16 + q) * ND + kbA;
  const bf16_t* brow[4];
#pragma unroll
  for (int j = 0; j < 4; j++)
    brow[j] = WbVt + ((size_t)b * NM + tn * 64 + j * 16 + q) * ND + kbB;

  v8f acc[4] = {};
  gemm_core_pipe(arow, brow, acc);

  int l0 = tm * 16 + hi * 8;
#pragma unroll
  for (int j = 0; j < 4; j++) {
    int m = tn * 64 + j * 16 + q;
    uint4 pk;
    pk.x = pack2(acc[j][0], acc[j][1]);
    pk.y = pack2(acc[j][2], acc[j][3]);
    pk.z = pack2(acc[j][4], acc[j][5]);
    pk.w = pack2(acc[j][6], acc[j][7]);
    *(uint4*)(Cml + ((size_t)b * NM + m) * NL + l0) = pk;   // C^T[m][l]
#pragma unroll
    for (int i = 0; i < 8; i++)                              // C[l][m]
      Clm[((size_t)b * NL + l0 + i) * NM + m] = f32_to_bf16_bits(acc[j][i]);
  }
}

// ---- GEMM 3: thin projections  (W @ V  and  W @ Q^T), K padded to 32 -------
__global__ void coatt_gemm_proj(const bf16_t* __restrict__ Apad,   // [32, 512]
                                const bf16_t* __restrict__ Bt,     // [b][Nn][512]
                                long btStride, int Nn,
                                float* __restrict__ outF,          // [b][32][Nn]
                                bf16_t* __restrict__ outB) {
  const int wpB = 2 * (Nn / 64);
  int w = (blockIdx.x * blockDim.x + threadIdx.x) >> 5;
  int lane = threadIdx.x & 31;
  int b = w / wpB;
  if (b >= NB) return;
  int t = w % wpB;
  int tm = t / (Nn / 64);                          // k-row tile (0..1)
  int tn = t % (Nn / 64);
  int q = lane & 15, hi = lane >> 4;
  int kbA = hi * 16, kbB = hi * 16;

  const bf16_t* arow = Apad + (size_t)(tm * 16 + q) * ND + kbA;
  const bf16_t* Btb = Bt + (size_t)b * btStride;
  const bf16_t* brow[4];
#pragma unroll
  for (int j = 0; j < 4; j++)
    brow[j] = Btb + (size_t)(tn * 64 + j * 16 + q) * ND + kbB;

  v8f acc[4] = {};
  gemm_core_pipe(arow, brow, acc);

#pragma unroll
  for (int j = 0; j < 4; j++) {
    int n = tn * 64 + j * 16 + q;
#pragma unroll
    for (int i = 0; i < 8; i++) {
      int k = tm * 16 + hi * 8 + i;
      size_t idx = ((size_t)b * NKP + k) * Nn + n;
      outF[idx] = acc[j][i];
      outB[idx] = f32_to_bf16(acc[j][i]);
    }
  }
}

// ---- GEMM 4a: s_v[m] = sum_k whv[k]*tanh(WvV[k,m] + (WqQ@C)[k,m]) ----------
__global__ void coatt_attn_v(const bf16_t* __restrict__ WqQb,   // [b][32][L]
                             const bf16_t* __restrict__ Cml,    // [b][M][L]
                             const float* __restrict__ WvVf,    // [b][32][M]
                             const float* __restrict__ whv,     // [32]
                             float* __restrict__ sv) {          // [b][M]
  const int wpB = NM / 64;                         // 16 waves/batch
  int w = (blockIdx.x * blockDim.x + threadIdx.x) >> 5;
  int lane = threadIdx.x & 31;
  int b = w / wpB;
  if (b >= NB) return;
  int tn = w % wpB;
  int q = lane & 15, hi = lane >> 4;
  int kbA = hi * 16, kbB = hi * 16;

  const bf16_t* Ab = WqQb + (size_t)b * NKP * NL;
  const bf16_t* a0row = Ab + (size_t)q * NL + kbA;          // k rows 0..15
  const bf16_t* a1row = Ab + (size_t)(16 + q) * NL + kbA;   // k rows 16..31
  const bf16_t* brow[4];
#pragma unroll
  for (int j = 0; j < 4; j++)
    brow[j] = Cml + ((size_t)b * NM + tn * 64 + j * 16 + q) * NL + kbB;

  v8f acc0[4] = {}, acc1[4] = {};
  for (int k0 = 0; k0 < NL; k0 += 32) {
    v16bf a0 = fragA(a0row + k0);
    v16bf a1 = fragA(a1row + k0);
#pragma unroll
    for (int j = 0; j < 4; j++) {
      v16bf bb = fragB(brow[j] + k0);
      acc0[j] = wmma_bf16(a0, bb, acc0[j]);
      acc1[j] = wmma_bf16(a1, bb, acc1[j]);
    }
  }
  const float* Wvb = WvVf + (size_t)b * NKP * NM;
#pragma unroll
  for (int j = 0; j < 4; j++) {
    int n = tn * 64 + j * 16 + q;                  // m index
    float s = 0.f;
#pragma unroll
    for (int i = 0; i < 8; i++) {
      int k0r = hi * 8 + i;
      int k1r = 16 + hi * 8 + i;
      s += whv[k0r] * tanhf(Wvb[(size_t)k0r * NM + n] + acc0[j][i]);
      s += whv[k1r] * tanhf(Wvb[(size_t)k1r * NM + n] + acc1[j][i]);
    }
    s += __shfl_xor(s, 16);                        // join the two k-halves
    if (hi == 0) sv[(size_t)b * NM + n] = s;
  }
}

// ---- GEMM 4b: s_q[l] = sum_k whq[k]*tanh(WqQ[k,l] + (WvV@C^T)[k,l]) --------
__global__ void coatt_attn_q(const bf16_t* __restrict__ WvVb,   // [b][32][M]
                             const bf16_t* __restrict__ Clm,    // [b][L][M]
                             const float* __restrict__ WqQf,    // [b][32][L]
                             const float* __restrict__ whq,     // [32]
                             float* __restrict__ sq) {          // [b][L]
  const int wpB = NL / 64;                         // 4 waves/batch
  int w = (blockIdx.x * blockDim.x + threadIdx.x) >> 5;
  int lane = threadIdx.x & 31;
  int b = w / wpB;
  if (b >= NB) return;
  int tn = w % wpB;
  int q = lane & 15, hi = lane >> 4;
  int kbA = hi * 16, kbB = hi * 16;

  const bf16_t* Ab = WvVb + (size_t)b * NKP * NM;
  const bf16_t* a0row = Ab + (size_t)q * NM + kbA;
  const bf16_t* a1row = Ab + (size_t)(16 + q) * NM + kbA;
  const bf16_t* brow[4];
#pragma unroll
  for (int j = 0; j < 4; j++)
    brow[j] = Clm + ((size_t)b * NL + tn * 64 + j * 16 + q) * NM + kbB;

  v8f acc0[4] = {}, acc1[4] = {};
  for (int k0 = 0; k0 < NM; k0 += 32) {
    v16bf a0 = fragA(a0row + k0);
    v16bf a1 = fragA(a1row + k0);
#pragma unroll
    for (int j = 0; j < 4; j++) {
      v16bf bb = fragB(brow[j] + k0);
      acc0[j] = wmma_bf16(a0, bb, acc0[j]);
      acc1[j] = wmma_bf16(a1, bb, acc1[j]);
    }
  }
  const float* Wqb = WqQf + (size_t)b * NKP * NL;
#pragma unroll
  for (int j = 0; j < 4; j++) {
    int n = tn * 64 + j * 16 + q;                  // l index
    float s = 0.f;
#pragma unroll
    for (int i = 0; i < 8; i++) {
      int k0r = hi * 8 + i;
      int k1r = 16 + hi * 8 + i;
      s += whq[k0r] * tanhf(Wqb[(size_t)k0r * NL + n] + acc0[j][i]);
      s += whq[k1r] * tanhf(Wqb[(size_t)k1r * NL + n] + acc1[j][i]);
    }
    s += __shfl_xor(s, 16);
    if (hi == 0) sq[(size_t)b * NL + n] = s;
  }
}

// ---- softmax over a row (one wave per batch) -------------------------------
__global__ void coatt_softmax(const float* __restrict__ s, float* __restrict__ a,
                              int n) {
  int b = blockIdx.x;
  int lane = threadIdx.x;
  const float* x = s + (size_t)b * n;
  float mx = -1e30f;
  for (int i = lane; i < n; i += 32) mx = fmaxf(mx, x[i]);
  for (int off = 16; off; off >>= 1) mx = fmaxf(mx, __shfl_xor(mx, off));
  float sum = 0.f;
  for (int i = lane; i < n; i += 32) sum += expf(x[i] - mx);
  for (int off = 16; off; off >>= 1) sum += __shfl_xor(sum, off);
  float inv = 1.f / sum;
  float* y = a + (size_t)b * n;
  for (int i = lane; i < n; i += 32) y[i] = expf(x[i] - mx) * inv;
}

// ---- pooling ---------------------------------------------------------------
__global__ void coatt_pool_v(const float* __restrict__ V,
                             const float* __restrict__ av,
                             float* __restrict__ v1) {
  int w = (blockIdx.x * blockDim.x + threadIdx.x) >> 5;   // wave -> (b, dcol)
  int lane = threadIdx.x & 31;
  int b = w >> 9;
  if (b >= NB) return;
  int dcol = w & (ND - 1);
  const float* row = V + ((size_t)b * ND + dcol) * NM;
  const float* a = av + (size_t)b * NM;
  float s = 0.f;
  for (int m = lane; m < NM; m += 32) s += row[m] * a[m];
  for (int off = 16; off; off >>= 1) s += __shfl_xor(s, off);
  if (!lane) v1[(size_t)b * ND + dcol] = s;
}

__global__ void coatt_pool_q(const float* __restrict__ Q,
                             const float* __restrict__ aq,
                             float* __restrict__ q1) {
  int b = blockIdx.x;
  int d = threadIdx.x;                               // 0..511
  const float* Qb = Q + (size_t)b * NL * ND;
  const float* a = aq + (size_t)b * NL;
  float s = 0.f;
  for (int l = 0; l < NL; l++) s += a[l] * Qb[(size_t)l * ND + d];
  q1[(size_t)b * ND + d] = s;
}

// ---- broadcast pooled vectors into the big outputs -------------------------
__global__ void coatt_broadcast(const float* __restrict__ v1,
                                const float* __restrict__ q1,
                                float4* __restrict__ out) {
  const size_t nv4 = (size_t)NB * NM * ND / 4;
  const size_t nq4 = (size_t)NB * NL * ND / 4;
  const size_t total = nv4 + nq4;
  for (size_t i = blockIdx.x * (size_t)blockDim.x + threadIdx.x; i < total;
       i += gridDim.x * (size_t)blockDim.x) {
    if (i < nv4) {
      size_t f = i * 4;
      int b = (int)(f / ((size_t)NM * ND));
      int dcol = (int)(f & (ND - 1));
      out[i] = *(const float4*)(v1 + (size_t)b * ND + dcol);
    } else {
      size_t f = (i - nv4) * 4;
      int b = (int)(f / ((size_t)NL * ND));
      int dcol = (int)(f & (ND - 1));
      out[i] = *(const float4*)(q1 + (size_t)b * ND + dcol);
    }
  }
}

// ---------------------------------------------------------------------------
extern "C" void kernel_launch(void* const* d_in, const int* in_sizes, int n_in,
                              void* d_out, int out_size, void* d_ws, size_t ws_size,
                              hipStream_t stream) {
  (void)in_sizes; (void)n_in; (void)out_size; (void)ws_size;
  const float* V   = (const float*)d_in[0];
  const float* Qf  = (const float*)d_in[1];
  const float* Wb  = (const float*)d_in[2];
  const float* Wv  = (const float*)d_in[3];
  const float* Wq  = (const float*)d_in[4];
  const float* whv = (const float*)d_in[5];
  const float* whq = (const float*)d_in[6];
  float* out = (float*)d_out;

  char* ws = (char*)d_ws;
  size_t off = 0;
  auto take = [&](size_t bytes) -> char* {
    char* p = ws + off;
    off += (bytes + 255) & ~(size_t)255;
    return p;
  };
  bf16_t* Vt    = (bf16_t*)take((size_t)NB * NM * ND * 2);
  bf16_t* Qb    = (bf16_t*)take((size_t)NB * NL * ND * 2);
  bf16_t* Wbb   = (bf16_t*)take((size_t)ND * ND * 2);
  bf16_t* Wvp   = (bf16_t*)take((size_t)NKP * ND * 2);
  bf16_t* Wqp   = (bf16_t*)take((size_t)NKP * ND * 2);
  float*  whvp  = (float*)take(NKP * 4);
  float*  whqp  = (float*)take(NKP * 4);
  bf16_t* WbVt  = (bf16_t*)take((size_t)NB * NM * ND * 2);
  bf16_t* Cml   = (bf16_t*)take((size_t)NB * NM * NL * 2);
  unsigned short* Clm = (unsigned short*)take((size_t)NB * NL * NM * 2);
  float*  WvVf  = (float*)take((size_t)NB * NKP * NM * 4);
  bf16_t* WvVb  = (bf16_t*)take((size_t)NB * NKP * NM * 2);
  float*  WqQf  = (float*)take((size_t)NB * NKP * NL * 4);
  bf16_t* WqQb  = (bf16_t*)take((size_t)NB * NKP * NL * 2);
  float*  sv    = (float*)take((size_t)NB * NM * 4);
  float*  sq    = (float*)take((size_t)NB * NL * 4);
  float*  av    = (float*)take((size_t)NB * NM * 4);
  float*  aq    = (float*)take((size_t)NB * NL * 4);
  float*  v1    = (float*)take((size_t)NB * ND * 4);
  float*  q1    = (float*)take((size_t)NB * ND * 4);

  // 0) conversions / layouts
  coatt_convert_bf16<<<2048, 256, 0, stream>>>(Qf, Qb, NB * NL * ND);
  coatt_convert_bf16<<<512, 256, 0, stream>>>(Wb, Wbb, ND * ND);
  coatt_pad_weights<<<(NKP * ND + 255) / 256, 256, 0, stream>>>(
      Wv, Wq, whv, whq, Wvp, Wqp, whvp, whqp);
  coatt_transpose_V<<<dim3(NM / 32, ND / 32, NB), dim3(32, 8), 0, stream>>>(V, Vt);

  // 1) WbV = W_b @ V  (stored transposed)
  coatt_gemm_wbv<<<(NB * 512) / 8, 256, 0, stream>>>(Wbb, Vt, WbVt);
  // 2) C = Q @ WbV  (both layouts)
  coatt_gemm_affinity<<<(NB * 256) / 8, 256, 0, stream>>>(Qb, WbVt, Cml, Clm);
  // 3) thin projections
  coatt_gemm_proj<<<(NB * 32) / 8, 256, 0, stream>>>(Wvp, Vt, (long)NM * ND, NM,
                                                     WvVf, WvVb);
  coatt_gemm_proj<<<(NB * 8) / 8, 256, 0, stream>>>(Wqp, Qb, (long)NL * ND, NL,
                                                    WqQf, WqQb);
  // 4) fused H GEMM + tanh + head-weighted reduction
  coatt_attn_v<<<(NB * 16) / 8, 256, 0, stream>>>(WqQb, Cml, WvVf, whvp, sv);
  coatt_attn_q<<<(NB * 4) / 8, 256, 0, stream>>>(WvVb, (const bf16_t*)Clm, WqQf,
                                                 whqp, sq);
  // 5) softmax
  coatt_softmax<<<NB, 32, 0, stream>>>(sv, av, NM);
  coatt_softmax<<<NB, 32, 0, stream>>>(sq, aq, NL);
  // 6) pooling
  coatt_pool_v<<<(NB * ND) / 8, 256, 0, stream>>>(V, av, v1);
  coatt_pool_q<<<NB, ND, 0, stream>>>(Qf, aq, q1);
  // 7) broadcast to outputs
  coatt_broadcast<<<8192, 256, 0, stream>>>(v1, q1, (float4*)out);
}
